// ChessNetwork_55413668053089
// MI455X (gfx1250) — matrix operand order repaired and checked by reading, not compiled, not run on hardware
//
#include <hip/hip_runtime.h>
#include <hip/hip_bf16.h>

typedef __bf16 bf16;
typedef __attribute__((ext_vector_type(16))) __bf16 v16bf;
typedef __attribute__((ext_vector_type(8)))  __bf16 v8bf;
typedef __attribute__((ext_vector_type(4)))  __bf16 v4bf;
typedef __attribute__((ext_vector_type(8)))  float  v8f;

#define NSQ   131072
#define NPC   49152
#define BATCH 2048
#define HDIM  128
#define DDIM  256
#define TDIM  128
#define ADIM  4672
#define E_ADJ 860160L
#define E_OCC 49152L
#define E_ATT 81920L
#define E_DEF 81920L
#define E_REV 49152L

__device__ __forceinline__ float gelu_f(float x) {
    const float c = 0.7978845608028654f;
    return 0.5f * x * (1.0f + tanhf(c * (x + 0.044715f * x * x * x)));
}

__device__ __forceinline__ void atomAddF(float* p, float v) {
    unsafeAtomicAdd(p, v);
}

// ---------------- weight transpose + f32->bf16 : dst is N x K (column-major of src K x N)
__global__ void transpose_cvt(const float* __restrict__ src, bf16* __restrict__ dst, int K, int N) {
    long tid = (long)blockIdx.x * blockDim.x + threadIdx.x;
    if (tid >= (long)K * N) return;
    int n = (int)(tid / K), k = (int)(tid % K);
    dst[(long)n * K + k] = (bf16)src[(long)k * N + n];
}

__global__ void concat_bias(const float* __restrict__ a, const float* __restrict__ b, float* __restrict__ o) {
    int i = threadIdx.x;                 // 256 threads
    o[i] = (i < 128) ? a[i] : b[i - 128];
}

__global__ void f32_to_bf16(const float* __restrict__ s, bf16* __restrict__ d, long n) {
    long i = (long)blockIdx.x * blockDim.x + threadIdx.x;
    if (i < n) d[i] = (bf16)s[i];
}

// ---------------- input embedding: h = gelu(x @ W_in + b), write f32 + bf16 copies
__global__ void __launch_bounds__(128) input_embed(
    const float* __restrict__ x_sq, const float* __restrict__ x_pc,
    const float* __restrict__ Wsq, const float* __restrict__ bsq,
    const float* __restrict__ Wpc, const float* __restrict__ bpc,
    float* __restrict__ hsf, bf16* __restrict__ hsb,
    float* __restrict__ hpf, bf16* __restrict__ hpb)
{
    int row = blockIdx.x;
    int t = threadIdx.x;                 // 0..127
    const float *x, *W, *b;
    float* hf; bf16* hb; long r;
    if (row < NSQ) { r = row;       x = x_sq + r * 16; W = Wsq; b = bsq; hf = hsf; hb = hsb; }
    else           { r = row - NSQ; x = x_pc + r * 16; W = Wpc; b = bpc; hf = hpf; hb = hpb; }
    float acc = b[t];
    #pragma unroll
    for (int f = 0; f < 16; ++f) acc += x[f] * W[f * 128 + t];
    float g = gelu_f(acc);
    hf[r * 128 + t] = g;
    hb[r * 128 + t] = (bf16)g;
}

// ---------------- generic bf16 WMMA GEMM: C = act(A(MxK,lda) @ Bt(NxK)^T + bias)
// OUT: 0 = store f32, 1 = store bf16, 2 = atomic segment-mean into seg (B x ldc)
template<bool ACT, int OUT>
__global__ void __launch_bounds__(128) wmma_gemm(
    const bf16* __restrict__ A, int lda,
    const bf16* __restrict__ Bt, int K,
    const float* __restrict__ bias,
    float* __restrict__ Cf, bf16* __restrict__ Cb, int ldc,
    float* __restrict__ seg, int rps, float sscale)
{
    const int lane = threadIdx.x & 31;
    const int wave = threadIdx.x >> 5;
    const int m16  = lane & 15;
    const int hh   = lane >> 4;
    const long row0 = (long)blockIdx.y * 64 + wave * 16;
    const long col0 = (long)blockIdx.x * 64;

    v8f acc[4] = {};
    const bf16* arow  = A  + (row0 + m16) * (long)lda;
    const bf16* bcol0 = Bt + (col0 + m16) * (long)K + hh * 16;

    for (int kk = 0; kk < K; kk += 32) {
        // A frag (16x32 bf16): lanes 0-15 hold K[0..15], lanes 16-31 K[16..31]
        v8bf lo = *(const v8bf*)(arow + kk + hh * 16);
        v8bf hi = *(const v8bf*)(arow + kk + hh * 16 + 8);
        v16bf a = __builtin_shufflevector(lo, hi, 0,1,2,3,4,5,6,7,8,9,10,11,12,13,14,15);
        #pragma unroll
        for (int t = 0; t < 4; ++t) {
            // B frag: column (col0+16t+m16), contiguous 16 K-values per lane half
            v16bf b = *(const v16bf*)(bcol0 + (long)t * 16 * K + kk);
            acc[t] = __builtin_amdgcn_wmma_f32_16x16x32_bf16(
                false, a, false, b, (short)0, acc[t], false, false);
        }
    }

    #pragma unroll
    for (int t = 0; t < 4; ++t) {
        long col = col0 + t * 16 + m16;
        float bv = bias ? bias[col] : 0.0f;
        #pragma unroll
        for (int i = 0; i < 8; ++i) {
            long row = row0 + hh * 8 + i;
            float v = acc[t][i] + bv;
            if (ACT) v = gelu_f(v);
            if (OUT == 0)      Cf[row * (long)ldc + col] = v;
            else if (OUT == 1) Cb[row * (long)ldc + col] = (bf16)v;
            else               atomAddF(seg + (row / rps) * (long)ldc + col, v * sscale);
        }
    }
}

// ---------------- edge scatter: agg[dst] += T[src][toff:toff+128]
__global__ void __launch_bounds__(256) scatter_add(
    const int* __restrict__ ei, long E,
    const bf16* __restrict__ T, int tstride, int toff,
    float* __restrict__ agg)
{
    long tid = (long)blockIdx.x * blockDim.x + threadIdx.x;
    long e = tid >> 5;
    if (e >= E) return;
    int q = (int)(tid & 31);
    long src = ei[e];
    long dst = ei[E + e];
    v4bf tv = *(const v4bf*)(T + src * tstride + toff + q * 4);
    float* base = agg + dst * 128 + q * 4;
    #pragma unroll
    for (int i = 0; i < 4; ++i) atomAddF(base + i, (float)tv[i]);
}

// ---------------- residual + gelu -> bf16 (overwrites h bf16 buffer)
__global__ void __launch_bounds__(256) residual_gelu(
    const float* __restrict__ h, const float* __restrict__ agg, bf16* __restrict__ h2, long n)
{
    long i = (long)blockIdx.x * blockDim.x + threadIdx.x;
    if (i < n) h2[i] = (bf16)gelu_f(h[i] + agg[i]);
}

// ---------------- value head: tanh(v1 @ W_vh + b_vh)
__global__ void __launch_bounds__(256) value_head(
    const bf16* __restrict__ p1v1, const float* __restrict__ Wvh,
    const float* __restrict__ bvh, float* __restrict__ out)
{
    int row = blockIdx.x * blockDim.x + threadIdx.x;
    if (row >= BATCH) return;
    float acc = bvh[0];
    const bf16* v = p1v1 + (long)row * 256 + 128;
    #pragma unroll 8
    for (int k = 0; k < 128; ++k) acc += (float)v[k] * Wvh[k];
    out[row] = tanhf(acc);
}

extern "C" void kernel_launch(void* const* d_in, const int* in_sizes, int n_in,
                              void* d_out, int out_size, void* d_ws, size_t ws_size,
                              hipStream_t stream)
{
    const float* x_sq    = (const float*)d_in[0];
    const float* x_pc    = (const float*)d_in[1];
    const float* W_in_sq = (const float*)d_in[2];
    const float* b_in_sq = (const float*)d_in[3];
    const float* W_in_pc = (const float*)d_in[4];
    const float* b_in_pc = (const float*)d_in[5];
    const float* W_adj   = (const float*)d_in[6];
    const float* W_occ   = (const float*)d_in[7];
    const float* W_att   = (const float*)d_in[8];
    const float* W_def   = (const float*)d_in[9];
    const float* W_rev   = (const float*)d_in[10];
    const float* W_out   = (const float*)d_in[11];
    const float* b_out   = (const float*)d_in[12];
    const float* W_pt    = (const float*)d_in[13];
    const float* b_pt    = (const float*)d_in[14];
    const float* W_vt    = (const float*)d_in[15];
    const float* b_vt    = (const float*)d_in[16];
    const float* W_ph    = (const float*)d_in[17];
    const float* b_ph    = (const float*)d_in[18];
    const float* W_vh    = (const float*)d_in[19];
    const float* b_vh    = (const float*)d_in[20];
    const int*   ei_adj  = (const int*)d_in[21];
    const int*   ei_occ  = (const int*)d_in[22];
    const int*   ei_att  = (const int*)d_in[23];
    const int*   ei_def  = (const int*)d_in[24];
    const int*   ei_rev  = (const int*)d_in[25];

    // ---- workspace carve-up (aggs + shared are contiguous for one memset) ----
    size_t off = 0;
    auto carve = [&](size_t bytes) -> void* {
        void* p = (char*)d_ws + off;
        off += (bytes + 255) & ~(size_t)255;
        return p;
    };
    float* h_sq_f  = (float*)carve((size_t)NSQ * 128 * 4);
    float* h_pc_f  = (float*)carve((size_t)NPC * 128 * 4);
    bf16*  h_sq_b  = (bf16*) carve((size_t)NSQ * 128 * 2);
    bf16*  h_pc_b  = (bf16*) carve((size_t)NPC * 128 * 2);
    bf16*  t_sq    = (bf16*) carve((size_t)NSQ * 256 * 2);   // [W_adj | W_rev]
    bf16*  t_pc    = (bf16*) carve((size_t)NPC * 384 * 2);   // [W_occ | W_att | W_def]
    float* agg_sq  = (float*)carve((size_t)NSQ * 128 * 4);
    float* agg_pc  = (float*)carve((size_t)NPC * 128 * 4);
    float* shared  = (float*)carve((size_t)BATCH * 256 * 4);
    bf16*  shared_b= (bf16*) carve((size_t)BATCH * 256 * 2);
    bf16*  p1v1    = (bf16*) carve((size_t)BATCH * 256 * 2); // [gelu(pt) | gelu(vt)]
    bf16*  WsqT    = (bf16*) carve((size_t)256 * 128 * 2);
    bf16*  WpcT    = (bf16*) carve((size_t)384 * 128 * 2);
    bf16*  WoutT   = (bf16*) carve((size_t)256 * 128 * 2);
    bf16*  WptvtT  = (bf16*) carve((size_t)256 * 256 * 2);
    bf16*  WphT    = (bf16*) carve((size_t)ADIM * 128 * 2);
    float* bias_cat= (float*)carve(256 * 4);

    float* policy_out = (float*)d_out;                         // 2048 x 4672
    float* value_out  = (float*)d_out + (size_t)BATCH * ADIM;  // 2048

    // ---- zero accumulators (agg_sq, agg_pc, shared are contiguous) ----
    hipMemsetAsync(agg_sq, 0, ((size_t)NSQ * 128 + (size_t)NPC * 128 + (size_t)BATCH * 256) * 4, stream);

    // ---- weights -> transposed bf16 (N x K) ----
    auto tcvt = [&](const float* s, bf16* d, int K, int N) {
        long n = (long)K * N;
        transpose_cvt<<<(int)((n + 255) / 256), 256, 0, stream>>>(s, d, K, N);
    };
    tcvt(W_adj, WsqT,                128, 128);
    tcvt(W_rev, WsqT + 128 * 128,    128, 128);
    tcvt(W_occ, WpcT,                128, 128);
    tcvt(W_att, WpcT + 128 * 128,    128, 128);
    tcvt(W_def, WpcT + 256 * 128,    128, 128);
    tcvt(W_out, WoutT,               128, 256);
    tcvt(W_pt,  WptvtT,              256, 128);
    tcvt(W_vt,  WptvtT + 128 * 256,  256, 128);
    tcvt(W_ph,  WphT,                128, ADIM);
    concat_bias<<<1, 256, 0, stream>>>(b_pt, b_vt, bias_cat);

    // ---- input embeddings ----
    input_embed<<<NSQ + NPC, 128, 0, stream>>>(x_sq, x_pc, W_in_sq, b_in_sq, W_in_pc, b_in_pc,
                                               h_sq_f, h_sq_b, h_pc_f, h_pc_b);

    // ---- edge transforms (dense GEMMs instead of per-edge matmul) ----
    wmma_gemm<false, 1><<<dim3(256 / 64, NSQ / 64), 128, 0, stream>>>(
        h_sq_b, 128, WsqT, 128, nullptr, nullptr, t_sq, 256, nullptr, 1, 0.f);
    wmma_gemm<false, 1><<<dim3(384 / 64, NPC / 64), 128, 0, stream>>>(
        h_pc_b, 128, WpcT, 128, nullptr, nullptr, t_pc, 384, nullptr, 1, 0.f);

    // ---- scatters ----
    auto scat = [&](const int* ei, long E, const bf16* T, int ts, int toff, float* agg) {
        long threads = E * 32;
        scatter_add<<<(int)((threads + 255) / 256), 256, 0, stream>>>(ei, E, T, ts, toff, agg);
    };
    scat(ei_adj, E_ADJ, t_sq, 256, 0,   agg_sq);   // h_sq @ W_adj  -> squares
    scat(ei_occ, E_OCC, t_pc, 384, 0,   agg_sq);   // h_pc @ W_occ  -> squares
    scat(ei_att, E_ATT, t_pc, 384, 128, agg_pc);   // h_pc @ W_att  -> pieces
    scat(ei_def, E_DEF, t_pc, 384, 256, agg_pc);   // h_pc @ W_def  -> pieces
    scat(ei_rev, E_REV, t_sq, 256, 128, agg_pc);   // h_sq @ W_rev  -> pieces

    // ---- residual + gelu (h2 overwrites bf16 h buffers) ----
    residual_gelu<<<(int)(((long)NSQ * 128 + 255) / 256), 256, 0, stream>>>(h_sq_f, agg_sq, h_sq_b, (long)NSQ * 128);
    residual_gelu<<<(int)(((long)NPC * 128 + 255) / 256), 256, 0, stream>>>(h_pc_f, agg_pc, h_pc_b, (long)NPC * 128);

    // ---- output GEMMs fused with segment-mean into shared (rows contiguous per batch) ----
    wmma_gemm<true, 2><<<dim3(256 / 64, NSQ / 64), 128, 0, stream>>>(
        h_sq_b, 128, WoutT, 128, b_out, nullptr, nullptr, 256, shared, 64, 1.0f / 64.0f);
    wmma_gemm<true, 2><<<dim3(256 / 64, NPC / 64), 128, 0, stream>>>(
        h_pc_b, 128, WoutT, 128, b_out, nullptr, nullptr, 256, shared, 24, 1.0f / 24.0f);

    // ---- heads ----
    f32_to_bf16<<<(int)(((long)BATCH * 256 + 255) / 256), 256, 0, stream>>>(shared, shared_b, (long)BATCH * 256);
    // p1v1 = gelu(shared @ [W_pt|W_vt] + [b_pt|b_vt])   (2048x256 @ 256x256)
    wmma_gemm<true, 1><<<dim3(256 / 64, BATCH / 64), 128, 0, stream>>>(
        shared_b, 256, WptvtT, 256, bias_cat, nullptr, p1v1, 256, nullptr, 1, 0.f);
    // policy = p1 @ W_ph + b_ph   (2048x128 @ 128x4672), A strided inside p1v1
    wmma_gemm<false, 0><<<dim3(ADIM / 64, BATCH / 64), 128, 0, stream>>>(
        p1v1, 256, WphT, 128, b_ph, policy_out, nullptr, ADIM, nullptr, 1, 0.f);
    // value = tanh(v1 @ W_vh + b_vh)
    value_head<<<(BATCH + 255) / 256, 256, 0, stream>>>(p1v1, W_vh, b_vh, value_out);
}
